// DLFG_79817672229311
// MI455X (gfx1250) — matrix operand order repaired and checked by reading, not compiled, not run on hardware
//
#include <hip/hip_runtime.h>
#include <hip/hip_bf16.h>
#include <math.h>

typedef __attribute__((ext_vector_type(16))) _Float16 v16h;
typedef __attribute__((ext_vector_type(8)))  _Float16 v8h;
typedef __attribute__((ext_vector_type(4)))  _Float16 v4h;
typedef __attribute__((ext_vector_type(8)))  float    v8f;

#define B_BATCH   4096
#define MOVIES    20000
#define UINFO     32
#define HISTK     (UINFO + MOVIES)      /* 20032 */
#define FACTORS   64
#define LRELU_SLOPE 0.01f
#define BN_SCALE  0.9759000729485332f   /* 1/sqrt(1 + 0.05) */

// ---------------------------------------------------------------------------
// Fragment loaders -> f16 WMMA operands, wave32 layouts per CDNA5 ISA 7.12.2:
//   A (16x32, MxK): lane L: M=L&15, K={koff..koff+7, koff+16..koff+23},
//                   koff=(L>>4)*8.
//   B (32x16, KxN): lane L: N=L&15, K={koff..koff+7, koff+16..koff+23},
//                   koff=(L>>4)*8 (same K order as A).
// All loaders are UNGUARDED: callers clamp out-of-range tile columns to a
// valid row (results for clamped columns are never stored).
// f32 overloads convert at load (v_cvt_pk); f16 overloads are pure b128 loads.
// ---------------------------------------------------------------------------
__device__ __forceinline__ void cvt4(v16h& v, int base, float4 f) {
  v[base + 0] = (_Float16)f.x; v[base + 1] = (_Float16)f.y;
  v[base + 2] = (_Float16)f.z; v[base + 3] = (_Float16)f.w;
}

__device__ __forceinline__ v16h load_a_frag(const float* __restrict__ A, int lda,
                                            int row, int k0, int lane) {
  int koff = k0 + ((lane >> 4) << 3);
  const float* p = A + (size_t)row * lda + koff;
  const float4* p4 = (const float4*)p;
  const float4* q4 = (const float4*)(p + 16);
  v16h a;
  cvt4(a, 0, p4[0]); cvt4(a, 4, p4[1]);
  cvt4(a, 8, q4[0]); cvt4(a, 12, q4[1]);
  return a;
}

__device__ __forceinline__ v16h load_a_frag(const _Float16* __restrict__ A, int lda,
                                            int row, int k0, int lane) {
  int koff = k0 + ((lane >> 4) << 3);
  const _Float16* p = A + (size_t)row * lda + koff;
  v8h lo = *(const v8h*)p;          // 16B aligned
  v8h hi = *(const v8h*)(p + 16);
  v16h a;
  #pragma unroll
  for (int e = 0; e < 8; ++e) { a[e] = lo[e]; a[e + 8] = hi[e]; }
  return a;
}

__device__ __forceinline__ v16h load_b_frag(const float* __restrict__ W, int ldw,
                                            int col, int k0, int lane) {
  int kb = k0 + ((lane >> 4) << 3);
  const float4* p4 = (const float4*)(W + (size_t)col * ldw + kb);
  v16h b;
  cvt4(b, 0, p4[0]); cvt4(b, 4, p4[1]); cvt4(b, 8, p4[4]); cvt4(b, 12, p4[5]);
  return b;
}

__device__ __forceinline__ v16h load_b_frag(const _Float16* __restrict__ W, int ldw,
                                            int col, int k0, int lane) {
  int kb = k0 + ((lane >> 4) << 3);
  const _Float16* p = W + (size_t)col * ldw + kb;
  v8h lo = *(const v8h*)p;          // 16B aligned
  v8h hi = *(const v8h*)(p + 16);
  v16h b;
  #pragma unroll
  for (int e = 0; e < 8; ++e) { b[e] = lo[e]; b[e + 8] = hi[e]; }
  return b;
}

// ---------------------------------------------------------------------------
// Fused GEMM: Y[M,Nreal] = epi(A[M,K] @ W[Nreal,K]^T + bias)
//   * wave computes 16 x (NB*16): one A-frag feeds NB WMMAs per K-step.
//   * block = 8 waves = 8 consecutive M-tiles, same N-group (W broadcast).
//   * grid msuper-major, ngroup-minor: blocks sharing the same 128 A-rows are
//     adjacent and sweep K together -> A streamed ~once through L2.
//   * SPLIT (compile-time): peel one K=32 step from a second A buffer
//     (f32 fallback layer-1 concat; extInfo is exactly one WMMA step).
//   * Inner loop is branch-free: B columns pre-clamped, loads batched ahead
//     of the WMMA chain (independent accumulators per step).
// ---------------------------------------------------------------------------
enum { EPI_LRELU_BN = 0, EPI_LRELU = 1, EPI_TANH = 2 };

template <int EPI, int NB, bool SPLIT, typename TA, typename TB, typename TY>
__global__ __launch_bounds__(256) void wmma_mlp_gemm(
    const TA* __restrict__ A0, int lda0, int ksplit,
    const TA* __restrict__ A1, int lda1,
    const TB* __restrict__ W,  int ldw,
    const float* __restrict__ bias,
    const float* __restrict__ bn_g, const float* __restrict__ bn_b,
    TY* __restrict__ Y, int ldy,
    int nGroups, int Nreal, int K)
{
  const int lane   = threadIdx.x & 31;
  const int wave   = threadIdx.x >> 5;
  const int msuper = blockIdx.x / nGroups;
  const int ngroup = blockIdx.x - msuper * nGroups;
  const int tM     = msuper * 8 + wave;
  const int rowA   = tM * 16 + (lane & 15);
  const int colB   = ngroup * (NB * 16) + (lane & 15);

  // Clamped load columns: tiles past Nreal load row Nreal-1 (never stored).
  int colL[NB];
  #pragma unroll
  for (int j = 0; j < NB; ++j) {
    int cc = colB + j * 16;
    colL[j] = (cc < Nreal) ? cc : (Nreal - 1);
  }

  v8f c[NB] = {};

  int kbeg = 0;
  if (SPLIT) {                        // layer-1 concat: one K=32 step from A0
    v16h bfr[NB];
    #pragma unroll
    for (int j = 0; j < NB; ++j) bfr[j] = load_b_frag(W, ldw, colL[j], 0, lane);
    v16h a = load_a_frag(A0, lda0, rowA, 0, lane);
    #pragma unroll
    for (int j = 0; j < NB; ++j)
      c[j] = __builtin_amdgcn_wmma_f32_16x16x32_f16(false, a, false, bfr[j],
                                                    (short)0, c[j], false, false);
    kbeg = ksplit;
  }

  for (int k0 = kbeg; k0 < K; k0 += 32) {
    v16h bfr[NB];
    #pragma unroll
    for (int j = 0; j < NB; ++j) bfr[j] = load_b_frag(W, ldw, colL[j], k0, lane);
    v16h a = load_a_frag(A1, lda1, rowA, SPLIT ? (k0 - ksplit) : k0, lane);
    #pragma unroll
    for (int j = 0; j < NB; ++j)
      c[j] = __builtin_amdgcn_wmma_f32_16x16x32_f16(false, a, false, bfr[j],
                                                    (short)0, c[j], false, false);
  }

  // C/D layout: VGPR r, lane l -> N = l&15 (in-tile), M = r + (l>>4)*8
  const int mbase = tM * 16 + ((lane >> 4) << 3);
  #pragma unroll
  for (int j = 0; j < NB; ++j) {
    const int n = colB + j * 16;
    if (n < Nreal) {
      const float bv = bias[n];
      const float g  = (EPI == EPI_LRELU_BN) ? bn_g[n] : 0.f;
      const float bb = (EPI == EPI_LRELU_BN) ? bn_b[n] : 0.f;
      #pragma unroll
      for (int r = 0; r < 8; ++r) {
        float v = c[j][r] + bv;
        if (EPI == EPI_TANH) {
          v = tanhf(v);
        } else {
          v = (v >= 0.f) ? v : LRELU_SLOPE * v;
          if (EPI == EPI_LRELU_BN) v = g * (v * BN_SCALE) + bb;
        }
        Y[(size_t)(mbase + r) * ldy + n] = (TY)v;
      }
    }
  }
}

// ---------------------------------------------------------------------------
// Final: out[b,m] = gen[b,0:64] . mf[m,0:64] + meanV[b] + gen[b,64] + mbias[m]
// K=64 -> two WMMA steps; N = 1250 exact 16-tiles (no guards anywhere).
// Write-bound: NT stores keep L2 free for the endlessly re-read gen/mf.
// ---------------------------------------------------------------------------
template <typename TB>
__global__ __launch_bounds__(256) void wmma_final(
    const float* __restrict__ gen,    // [B, 80]
    const TB*    __restrict__ mf,     // [MOVIES, 64]
    const float* __restrict__ mbias,  // [MOVIES]
    const float* __restrict__ meanV,  // [B]
    float* __restrict__ out)          // [B, MOVIES]
{
  const int tilesM = B_BATCH / 16;    // 256
  const int lane = threadIdx.x & 31;
  const int wave = threadIdx.x >> 5;
  const int tile = blockIdx.x * 8 + wave;   // grid exact: 1250*256/8 blocks
  const int tN = tile / tilesM;
  const int tM = tile - tN * tilesM;
  const int rowA = tM * 16 + (lane & 15);
  const int colW = tN * 16 + (lane & 15);

  v8f c = {};
  #pragma unroll
  for (int s = 0; s < 2; ++s) {
    v16h b = load_b_frag(mf, 64, colW, s * 32, lane);
    v16h a = load_a_frag(gen, 80, rowA, s * 32, lane);
    c = __builtin_amdgcn_wmma_f32_16x16x32_f16(false, a, false, b,
                                               (short)0, c, false, false);
  }

  const int n  = tN * 16 + (lane & 15);
  const float mb = mbias[n];
  const int mbase = tM * 16 + ((lane >> 4) << 3);
  #pragma unroll
  for (int r = 0; r < 8; ++r) {
    const int b = mbase + r;
    const float add = meanV[b] + gen[(size_t)b * 80 + 64] + mb;
    __builtin_nontemporal_store(c[r] + add, &out[(size_t)b * MOVIES + n]);
  }
}

// ---------------------------------------------------------------------------
// Fast path: pack history (extInfo | ratings) to f16 AND compute the masked
// row statistics in the same single pass over ratings.
// ---------------------------------------------------------------------------
__global__ __launch_bounds__(256) void hist_pack_stats(
    const float* __restrict__ extInfo, const float* __restrict__ ratings,
    _Float16* __restrict__ hist, float* __restrict__ s, float* __restrict__ cnt)
{
  __shared__ float ss[256], sc[256];
  const int b = blockIdx.x;
  const float* row = ratings + (size_t)b * MOVIES;
  _Float16* hrow = hist + (size_t)b * HISTK;
  if (threadIdx.x < UINFO)
    hrow[threadIdx.x] = (_Float16)extInfo[b * UINFO + threadIdx.x];
  float ls = 0.f, lc = 0.f;
  for (int j = threadIdx.x; j < MOVIES; j += 256) {
    float r = __builtin_nontemporal_load(&row[j]);
    hrow[UINFO + j] = (_Float16)r;
    if (r > 0.f) { ls += r; lc += 1.f; }
  }
  ss[threadIdx.x] = ls; sc[threadIdx.x] = lc;
  __syncthreads();
  for (int off = 128; off > 0; off >>= 1) {
    if (threadIdx.x < off) {
      ss[threadIdx.x] += ss[threadIdx.x + off];
      sc[threadIdx.x] += sc[threadIdx.x + off];
    }
    __syncthreads();
  }
  if (threadIdx.x == 0) { s[b] = ss[0]; cnt[b] = sc[0]; }
}

// f32 -> f16 bulk conversion, 4 elements/thread (n must be a multiple of 4).
__global__ __launch_bounds__(256) void cvt_f32_f16(const float* __restrict__ in,
                                                   _Float16* __restrict__ out, int n4)
{
  int i = blockIdx.x * 256 + threadIdx.x;
  if (i < n4) {
    float4 f = ((const float4*)in)[i];
    v4h h;
    h[0] = (_Float16)f.x; h[1] = (_Float16)f.y;
    h[2] = (_Float16)f.z; h[3] = (_Float16)f.w;
    ((v4h*)out)[i] = h;
  }
}

// ---------------------------------------------------------------------------
// Fallback-path masked row statistics (no packing).
// ---------------------------------------------------------------------------
__global__ __launch_bounds__(256) void row_stats(const float* __restrict__ ratings,
                                                 float* __restrict__ s,
                                                 float* __restrict__ cnt)
{
  __shared__ float ss[256], sc[256];
  const int b = blockIdx.x;
  const float* row = ratings + (size_t)b * MOVIES;
  float ls = 0.f, lc = 0.f;
  for (int j = threadIdx.x; j < MOVIES; j += 256) {
    float r = row[j];
    if (r > 0.f) { ls += r; lc += 1.f; }
  }
  ss[threadIdx.x] = ls; sc[threadIdx.x] = lc;
  __syncthreads();
  for (int off = 128; off > 0; off >>= 1) {
    if (threadIdx.x < off) {
      ss[threadIdx.x] += ss[threadIdx.x + off];
      sc[threadIdx.x] += sc[threadIdx.x + off];
    }
    __syncthreads();
  }
  if (threadIdx.x == 0) { s[b] = ss[0]; cnt[b] = sc[0]; }
}

__global__ __launch_bounds__(256) void meanv_kernel(const float* __restrict__ s,
                                                    const float* __restrict__ cnt,
                                                    float* __restrict__ meanV)
{
  __shared__ float ss[256], sc[256];
  float ls = 0.f, lc = 0.f;
  for (int b = threadIdx.x; b < B_BATCH; b += 256) { ls += s[b]; lc += cnt[b]; }
  ss[threadIdx.x] = ls; sc[threadIdx.x] = lc;
  __syncthreads();
  for (int off = 128; off > 0; off >>= 1) {
    if (threadIdx.x < off) {
      ss[threadIdx.x] += ss[threadIdx.x + off];
      sc[threadIdx.x] += sc[threadIdx.x + off];
    }
    __syncthreads();
  }
  const float gm = ss[0] / sc[0];
  for (int b = threadIdx.x; b < B_BATCH; b += 256) {
    float cv = cnt[b];
    meanV[b] = (cv >= 10.f) ? (s[b] / fmaxf(cv, 1.f)) : gm;
  }
}

// ---------------------------------------------------------------------------
extern "C" void kernel_launch(void* const* d_in, const int* in_sizes, int n_in,
                              void* d_out, int out_size, void* d_ws, size_t ws_size,
                              hipStream_t stream)
{
  const float* extInfo = (const float*)d_in[0];
  const float* ratings = (const float*)d_in[1];
  const float* w1 = (const float*)d_in[2];   const float* b1 = (const float*)d_in[3];
  const float* w2 = (const float*)d_in[4];   const float* b2 = (const float*)d_in[5];
  const float* w3 = (const float*)d_in[6];   const float* b3 = (const float*)d_in[7];
  const float* w4 = (const float*)d_in[8];   const float* b4 = (const float*)d_in[9];
  const float* w5 = (const float*)d_in[10];  const float* b5 = (const float*)d_in[11];
  const float* w6 = (const float*)d_in[12];  const float* b6 = (const float*)d_in[13];
  const float* bn_g = (const float*)d_in[14];
  const float* bn_b = (const float*)d_in[15];
  const float* mf    = (const float*)d_in[16];
  const float* mbias = (const float*)d_in[17];
  float* out = (float*)d_out;

  // --- workspace arena (256B-aligned carving) ---
  char* base = (char*)d_ws;
  size_t off = 0;
  auto alloc = [&](size_t bytes) -> void* {
    void* p = base + off;
    off = (off + bytes + 255) & ~(size_t)255;
    return p;
  };

  // common region
  float* gen  = (float*)alloc((size_t)B_BATCH * 80 * sizeof(float));
  float* sArr = (float*)alloc(B_BATCH * sizeof(float));
  float* cArr = (float*)alloc(B_BATCH * sizeof(float));
  float* mV   = (float*)alloc(B_BATCH * sizeof(float));
  const size_t common_end = off;

  // f16 fast-path region
  _Float16* hist16 = (_Float16*)alloc((size_t)B_BATCH * HISTK * 2);
  _Float16* w1h    = (_Float16*)alloc((size_t)1024 * HISTK * 2);
  _Float16* w2h    = (_Float16*)alloc((size_t)512 * 1024 * 2);
  _Float16* w3h    = (_Float16*)alloc((size_t)256 * 512 * 2);
  _Float16* w4h    = (_Float16*)alloc((size_t)512 * 256 * 2);
  _Float16* w5h    = (_Float16*)alloc((size_t)1024 * 512 * 2);
  _Float16* w6h    = (_Float16*)alloc((size_t)65 * 1024 * 2);
  _Float16* mfh    = (_Float16*)alloc((size_t)MOVIES * FACTORS * 2);
  _Float16* y1h    = (_Float16*)alloc((size_t)B_BATCH * 1024 * 2);
  _Float16* y2h    = (_Float16*)alloc((size_t)B_BATCH * 512 * 2);
  _Float16* y3h    = (_Float16*)alloc((size_t)B_BATCH * 256 * 2);
  const size_t need_f16 = off;
  _Float16* y4h = y2h;   // aliases: dead by the time they are rewritten
  _Float16* y5h = y1h;

  const dim3 blk(256);
  const int MS = (B_BATCH / 16) / 8;                 // 32 M-superblocks
  auto cdiv = [](int a, int b) { return (a + b - 1) / b; };

  if (ws_size >= need_f16) {
    // ------------------------- f16 fast path -------------------------
    hist_pack_stats<<<B_BATCH, blk, 0, stream>>>(extInfo, ratings, hist16, sArr, cArr);
    meanv_kernel<<<1, blk, 0, stream>>>(sArr, cArr, mV);

    cvt_f32_f16<<<cdiv(1024 * HISTK / 4, 256), blk, 0, stream>>>(w1, w1h, 1024 * HISTK / 4);
    cvt_f32_f16<<<cdiv(512 * 1024 / 4, 256), blk, 0, stream>>>(w2, w2h, 512 * 1024 / 4);
    cvt_f32_f16<<<cdiv(256 * 512 / 4, 256), blk, 0, stream>>>(w3, w3h, 256 * 512 / 4);
    cvt_f32_f16<<<cdiv(512 * 256 / 4, 256), blk, 0, stream>>>(w4, w4h, 512 * 256 / 4);
    cvt_f32_f16<<<cdiv(1024 * 512 / 4, 256), blk, 0, stream>>>(w5, w5h, 1024 * 512 / 4);
    cvt_f32_f16<<<cdiv(65 * 1024 / 4, 256), blk, 0, stream>>>(w6, w6h, 65 * 1024 / 4);
    cvt_f32_f16<<<cdiv(MOVIES * FACTORS / 4, 256), blk, 0, stream>>>(mf, mfh, MOVIES * FACTORS / 4);

    wmma_mlp_gemm<EPI_LRELU_BN, 8, false, _Float16, _Float16, _Float16>
        <<<MS * 8, blk, 0, stream>>>(hist16, HISTK, 0, hist16, HISTK, w1h, HISTK,
                                     b1, bn_g, bn_b, y1h, 1024, 8, 1024, HISTK);
    wmma_mlp_gemm<EPI_LRELU, 4, false, _Float16, _Float16, _Float16>
        <<<MS * 8, blk, 0, stream>>>(y1h, 1024, 0, y1h, 1024, w2h, 1024,
                                     b2, nullptr, nullptr, y2h, 512, 8, 512, 1024);
    wmma_mlp_gemm<EPI_LRELU, 4, false, _Float16, _Float16, _Float16>
        <<<MS * 4, blk, 0, stream>>>(y2h, 512, 0, y2h, 512, w3h, 512,
                                     b3, nullptr, nullptr, y3h, 256, 4, 256, 512);
    wmma_mlp_gemm<EPI_LRELU, 4, false, _Float16, _Float16, _Float16>
        <<<MS * 8, blk, 0, stream>>>(y3h, 256, 0, y3h, 256, w4h, 256,
                                     b4, nullptr, nullptr, y4h, 512, 8, 512, 256);
    wmma_mlp_gemm<EPI_LRELU, 8, false, _Float16, _Float16, _Float16>
        <<<MS * 8, blk, 0, stream>>>(y4h, 512, 0, y4h, 512, w5h, 512,
                                     b5, nullptr, nullptr, y5h, 1024, 8, 1024, 512);
    wmma_mlp_gemm<EPI_TANH, 5, false, _Float16, _Float16, float>
        <<<MS * 1, blk, 0, stream>>>(y5h, 1024, 0, y5h, 1024, w6h, 1024,
                                     b6, nullptr, nullptr, gen, 80, 1, 65, 1024);

    wmma_final<_Float16><<<(1250 * 256) / 8, blk, 0, stream>>>(gen, mfh, mbias, mV, out);
  } else {
    // ------------------------- f32 fallback path -------------------------
    off = common_end;
    float* y1 = (float*)alloc((size_t)B_BATCH * 1024 * sizeof(float));
    float* y2 = (float*)alloc((size_t)B_BATCH * 512 * sizeof(float));
    float* y3 = (float*)alloc((size_t)B_BATCH * 256 * sizeof(float));
    float* y4 = y2;
    float* y5 = y1;

    row_stats<<<B_BATCH, blk, 0, stream>>>(ratings, sArr, cArr);
    meanv_kernel<<<1, blk, 0, stream>>>(sArr, cArr, mV);

    wmma_mlp_gemm<EPI_LRELU_BN, 8, true, float, float, float>
        <<<MS * 8, blk, 0, stream>>>(extInfo, UINFO, UINFO, ratings, MOVIES, w1, HISTK,
                                     b1, bn_g, bn_b, y1, 1024, 8, 1024, HISTK);
    wmma_mlp_gemm<EPI_LRELU, 4, false, float, float, float>
        <<<MS * 8, blk, 0, stream>>>(y1, 1024, 0, y1, 1024, w2, 1024,
                                     b2, nullptr, nullptr, y2, 512, 8, 512, 1024);
    wmma_mlp_gemm<EPI_LRELU, 4, false, float, float, float>
        <<<MS * 4, blk, 0, stream>>>(y2, 512, 0, y2, 512, w3, 512,
                                     b3, nullptr, nullptr, y3, 256, 4, 256, 512);
    wmma_mlp_gemm<EPI_LRELU, 4, false, float, float, float>
        <<<MS * 8, blk, 0, stream>>>(y3, 256, 0, y3, 256, w4, 256,
                                     b4, nullptr, nullptr, y4, 512, 8, 512, 256);
    wmma_mlp_gemm<EPI_LRELU, 8, false, float, float, float>
        <<<MS * 8, blk, 0, stream>>>(y4, 512, 0, y4, 512, w5, 512,
                                     b5, nullptr, nullptr, y5, 1024, 8, 1024, 512);
    wmma_mlp_gemm<EPI_TANH, 5, false, float, float, float>
        <<<MS * 1, blk, 0, stream>>>(y5, 1024, 0, y5, 1024, w6, 1024,
                                     b6, nullptr, nullptr, gen, 80, 1, 65, 1024);

    wmma_final<float><<<(1250 * 256) / 8, blk, 0, stream>>>(gen, mf, mbias, mV, out);
  }
}
